// TemporalModel_39728447488218
// MI455X (gfx1250) — compile-verified, matrix operand order checked by reading, not run on hardware
//
#include <hip/hip_runtime.h>
#include <hip/hip_bf16.h>

#define Bn   8
#define Ln   1024
#define DIN  5
#define Hn   512
#define NLn  4
#define En   1024
#define Nn   16
#define DCn  4
#define Rn   32
#define Mn   (Bn * Ln)   // 8192 token rows

typedef __bf16 bf16t;
typedef __attribute__((ext_vector_type(16))) __bf16 v16bf;
typedef __attribute__((ext_vector_type(8)))  __bf16 v8bf;
typedef __attribute__((ext_vector_type(8)))  float  v8f;

__device__ __forceinline__ bf16t f2bf(float f) {
  union { float f; unsigned u; } in; in.f = f;
  unsigned r = in.u + 0x7FFFu + ((in.u >> 16) & 1u);   // RNE
  union { unsigned short s; bf16t b; } out;
  out.s = (unsigned short)(r >> 16);
  return out.b;
}

__device__ __forceinline__ float siluf(float x) { return x / (1.0f + __expf(-x)); }
__device__ __forceinline__ float softplusf(float x) { return (x > 20.0f) ? x : log1pf(__expf(x)); }

// ---------------------------------------------------------------- converts (weights only)
__global__ void k_f32_to_bf16(const float* __restrict__ src, bf16t* __restrict__ dst, int n) {
  int i = blockIdx.x * blockDim.x + threadIdx.x;
  if (i < n) dst[i] = f2bf(src[i]);
}

// ------------------------------------------------- input projection (K=5), fused bf16 out
__global__ void k_inproj(const float* __restrict__ x, const float* __restrict__ w,
                         const float* __restrict__ bias, float* __restrict__ h,
                         bf16t* __restrict__ hbf) {
  int i = blockIdx.x * blockDim.x + threadIdx.x;   // over Mn*Hn
  int hh  = i & (Hn - 1);
  int row = i >> 9;                                // Hn == 512
  const float* xr = x + (size_t)row * DIN;
  const float* wr = w + (size_t)hh * DIN;
  float s = bias[hh];
#pragma unroll
  for (int d = 0; d < DIN; ++d) s += xr[d] * wr[d];
  h[i] = s;
  hbf[i] = f2bf(s);
}

// ------------------------------------------------- simple WMMA GEMM (small N or K)
// C[M x N] (f32, ldc) = A[M x K](bf16, lda) * B[N x K]^T(bf16, ldb)
// One wave per 16x16 tile; 8 waves/block; grid = (N/16, M/128).
__launch_bounds__(256)
__global__ void k_gemm_wmma(const bf16t* __restrict__ A, const bf16t* __restrict__ Bm,
                            float* __restrict__ C, bf16t* __restrict__ Cbf,
                            int K, int lda, int ldb, int ldc,
                            const float* __restrict__ bias, int act) {
  int lane = threadIdx.x & 31;
  int wave = threadIdx.x >> 5;
  int lo = lane & 15;
  int hi = lane >> 4;
  int tm = (blockIdx.y * 8 + wave) * 16;
  int tn = blockIdx.x * 16;
  const bf16t* ar = A  + (size_t)(tm + lo) * lda;
  const bf16t* br = Bm + (size_t)(tn + lo) * ldb;
  v8f acc = {};
  for (int k0 = 0; k0 < K; k0 += 32) {
    v16bf a, b;
    ((v8bf*)&a)[0] = *(const v8bf*)(ar + k0 + 8 * hi);
    ((v8bf*)&a)[1] = *(const v8bf*)(ar + k0 + 16 + 8 * hi);
    ((v8bf*)&b)[0] = *(const v8bf*)(br + k0 + 8 * hi);
    ((v8bf*)&b)[1] = *(const v8bf*)(br + k0 + 16 + 8 * hi);
    acc = __builtin_amdgcn_wmma_f32_16x16x32_bf16(false, a, false, b, (short)0, acc, false, false);
  }
  int col = tn + lo;
  float bv = bias ? bias[col] : 0.0f;
#pragma unroll
  for (int r = 0; r < 8; ++r) {
    int row = tm + r + 8 * hi;
    float v = acc[r] + bv;
    if (act == 1) v = softplusf(v);
    C[(size_t)row * ldc + col] = v;
    if (Cbf) Cbf[(size_t)row * ldc + col] = f2bf(v);
  }
}

// ------------------------------------------------- register-blocked WMMA GEMM
// 64x64 macro-tile per wave: 4 A frags x 4 B frags -> 16 WMMAs per k-step (4x reuse).
// grid = (N/64, M/512); 8 waves/block, each wave owns one macro-tile row slice.
__launch_bounds__(256)
__global__ void k_gemm_wmma_blk(const bf16t* __restrict__ A, const bf16t* __restrict__ Bm,
                                float* __restrict__ C, bf16t* __restrict__ Cbf,
                                int K, int lda, int ldb, int ldc) {
  int lane = threadIdx.x & 31;
  int wave = threadIdx.x >> 5;
  int lo = lane & 15;
  int hi = lane >> 4;
  int tm = (blockIdx.y * 8 + wave) * 64;
  int tn = blockIdx.x * 64;
  const bf16t* ar[4];
  const bf16t* br[4];
#pragma unroll
  for (int i = 0; i < 4; ++i) {
    ar[i] = A  + (size_t)(tm + 16 * i + lo) * lda;
    br[i] = Bm + (size_t)(tn + 16 * i + lo) * ldb;
  }
  v8f acc[16];
#pragma unroll
  for (int i = 0; i < 16; ++i) acc[i] = {};
  for (int k0 = 0; k0 < K; k0 += 32) {
    v16bf a[4], b[4];
#pragma unroll
    for (int i = 0; i < 4; ++i) {
      ((v8bf*)&a[i])[0] = *(const v8bf*)(ar[i] + k0 + 8 * hi);
      ((v8bf*)&a[i])[1] = *(const v8bf*)(ar[i] + k0 + 16 + 8 * hi);
      ((v8bf*)&b[i])[0] = *(const v8bf*)(br[i] + k0 + 8 * hi);
      ((v8bf*)&b[i])[1] = *(const v8bf*)(br[i] + k0 + 16 + 8 * hi);
    }
#pragma unroll
    for (int i = 0; i < 4; ++i)
#pragma unroll
      for (int j = 0; j < 4; ++j)
        acc[i * 4 + j] = __builtin_amdgcn_wmma_f32_16x16x32_bf16(
            false, a[i], false, b[j], (short)0, acc[i * 4 + j], false, false);
  }
#pragma unroll
  for (int i = 0; i < 4; ++i)
#pragma unroll
    for (int j = 0; j < 4; ++j) {
      int col = tn + 16 * j + lo;
#pragma unroll
      for (int r = 0; r < 8; ++r) {
        int row = tm + 16 * i + r + 8 * hi;
        float v = acc[i * 4 + j][r];
        C[(size_t)row * ldc + col] = v;
        if (Cbf) Cbf[(size_t)row * ldc + col] = f2bf(v);
      }
    }
}

// ------------------------------------------------- causal depthwise conv + SiLU (+bf16)
__global__ void k_conv_silu(const float* __restrict__ xz, const float* __restrict__ cw,
                            const float* __restrict__ cb, float* __restrict__ u,
                            bf16t* __restrict__ ubf) {
  int i = blockIdx.x * blockDim.x + threadIdx.x;    // over Mn*En
  int e   = i & (En - 1);
  int row = i >> 10;                                // En == 1024
  int t = row & (Ln - 1);
  const float* col = xz + (size_t)row * (2 * En) + e;
  float s = cb[e];
#pragma unroll
  for (int j = 0; j < DCn; ++j) {
    int tt = t - (DCn - 1) + j;
    if (tt >= 0) s += col[(j - (DCn - 1)) * (2 * En)] * cw[e * DCn + j];
  }
  float v = siluf(s);
  u[i] = v;
  ubf[i] = f2bf(v);
}

// ------------------------------------------------- selective scan (sequential in t)
#define TS 8
__launch_bounds__(256)
__global__ void k_scan(const float* __restrict__ dt, const float* __restrict__ dbc,
                       const float* __restrict__ xz, float* __restrict__ u,
                       bf16t* __restrict__ ybf,
                       const float* __restrict__ Alog, const float* __restrict__ Dv) {
  int b = blockIdx.x >> 2;
  int e = ((blockIdx.x & 3) << 8) + threadIdx.x;
  float a[Nn];
#pragma unroll
  for (int n = 0; n < Nn; ++n) a[n] = -__expf(Alog[e * Nn + n]);
  float dcoef = Dv[e];
  float hs[Nn];
#pragma unroll
  for (int n = 0; n < Nn; ++n) hs[n] = 0.0f;
  __shared__ float sBC[TS * 32];
  const size_t rowbase = (size_t)b * Ln;
  for (int t0 = 0; t0 < Ln; t0 += TS) {
    __syncthreads();
    {
      int i = threadIdx.x >> 5, j = threadIdx.x & 31;
      sBC[threadIdx.x] = dbc[(rowbase + t0 + i) * 64 + 32 + j];   // [B(16) | C(16)]
    }
    __syncthreads();
    for (int ti = 0; ti < TS; ++ti) {
      size_t off = (rowbase + t0 + ti) * En + e;
      float dtv = dt[off];
      float uv  = u[off];
      float zv  = xz[(rowbase + t0 + ti) * (2 * En) + En + e];
      const float* Bv = sBC + ti * 32;
      float dtu = dtv * uv;
      float y = 0.0f;
#pragma unroll
      for (int n = 0; n < Nn; ++n) {
        float dA = __expf(dtv * a[n]);
        hs[n] = dA * hs[n] + dtu * Bv[n];
        y += hs[n] * Bv[16 + n];
      }
      y = (y + uv * dcoef) * siluf(zv);
      u[off] = y;
      ybf[off] = f2bf(y);
    }
  }
}

// ------------------------------------------------- head
__global__ void k_head(const float* __restrict__ h, const int* __restrict__ lengths,
                       const float* __restrict__ ow, const float* __restrict__ ob,
                       float* __restrict__ out) {
  int b = blockIdx.x;
  int t = lengths[b] - 1;
  const float* row = h + ((size_t)b * Ln + t) * Hn;
  __shared__ float red[256];
  float s = 0.0f;
  for (int i = threadIdx.x; i < Hn; i += 256) s += row[i] * ow[i];
  red[threadIdx.x] = s;
  __syncthreads();
  for (int k = 128; k > 0; k >>= 1) {
    if ((int)threadIdx.x < k) red[threadIdx.x] += red[threadIdx.x + k];
    __syncthreads();
  }
  if (threadIdx.x == 0) out[b] = red[0] + ob[0];
}

// ----------------------------------------------------------------------
extern "C" void kernel_launch(void* const* d_in, const int* in_sizes, int n_in,
                              void* d_out, int out_size, void* d_ws, size_t ws_size,
                              hipStream_t stream) {
  (void)in_sizes; (void)n_in; (void)out_size; (void)ws_size;
  const float* x        = (const float*)d_in[0];
  const int*   lengths  = (const int*)  d_in[1];
  const float* in_w     = (const float*)d_in[2];
  const float* in_b     = (const float*)d_in[3];
  const float* l_inproj = (const float*)d_in[4];
  const float* l_convw  = (const float*)d_in[5];
  const float* l_convb  = (const float*)d_in[6];
  const float* l_xproj  = (const float*)d_in[7];
  const float* l_dtw    = (const float*)d_in[8];
  const float* l_dtb    = (const float*)d_in[9];
  const float* l_Alog   = (const float*)d_in[10];
  const float* l_D      = (const float*)d_in[11];
  const float* l_outw   = (const float*)d_in[12];
  const float* out_w    = (const float*)d_in[13];
  const float* out_b    = (const float*)d_in[14];
  float* out = (float*)d_out;

  char* p = (char*)d_ws;
  auto alloc = [&](size_t bytes) { char* r = p; p += (bytes + 255) & ~(size_t)255; return r; };
  float* xz      = (float*)alloc((size_t)Mn * 2 * En * 4);   // 64 MB
  float* u       = (float*)alloc((size_t)Mn * En * 4);       // 32 MB (u, later y)
  float* dtbuf   = (float*)alloc((size_t)Mn * En * 4);       // 32 MB
  float* dbc     = (float*)alloc((size_t)Mn * 64 * 4);       //  2 MB
  float* h       = (float*)alloc((size_t)Mn * Hn * 4);       // 16 MB
  bf16t* h_bf    = (bf16t*)alloc((size_t)Mn * Hn * 2);
  bf16t* u_bf    = (bf16t*)alloc((size_t)Mn * En * 2);
  bf16t* dbc_bf  = (bf16t*)alloc((size_t)Mn * 64 * 2);
  bf16t* w_in_bf = (bf16t*)alloc((size_t)2 * En * Hn * 2);
  bf16t* w_x_bf  = (bf16t*)alloc((size_t)64 * En * 2);
  bf16t* w_dt_bf = (bf16t*)alloc((size_t)En * Rn * 2);
  bf16t* w_ow_bf = (bf16t*)alloc((size_t)Hn * En * 2);

  auto cvt = [&](const float* s, bf16t* d, int n) {
    k_f32_to_bf16<<<(n + 255) / 256, 256, 0, stream>>>(s, d, n);
  };

  // input projection writes h and h_bf
  k_inproj<<<Mn * Hn / 256, 256, 0, stream>>>(x, in_w, in_b, h, h_bf);

  for (int i = 0; i < NLn; ++i) {
    const float* inproj = l_inproj + (size_t)i * 2 * En * Hn;
    const float* convw  = l_convw  + (size_t)i * En * DCn;
    const float* convb  = l_convb  + (size_t)i * En;
    const float* xproj  = l_xproj  + (size_t)i * 64 * En;
    const float* dtw    = l_dtw    + (size_t)i * En * Rn;
    const float* dtb    = l_dtb    + (size_t)i * En;
    const float* Alog   = l_Alog   + (size_t)i * En * Nn;
    const float* Dv     = l_D      + (size_t)i * En;
    const float* outw   = l_outw   + (size_t)i * Hn * En;

    cvt(inproj, w_in_bf, 2 * En * Hn);
    cvt(xproj,  w_x_bf,  64 * En);
    cvt(dtw,    w_dt_bf, En * Rn);
    cvt(outw,   w_ow_bf, Hn * En);

    // GEMM1 (blocked): xz[M,2E] = h @ inproj^T   (M=8192, N=2048, K=512)
    k_gemm_wmma_blk<<<dim3(2 * En / 64, Mn / 512), 256, 0, stream>>>(
        h_bf, w_in_bf, xz, nullptr, Hn, Hn, Hn, 2 * En);
    // causal depthwise conv + SiLU -> u, u_bf
    k_conv_silu<<<Mn * En / 256, 256, 0, stream>>>(xz, convw, convb, u, u_bf);
    // GEMM2 (simple, N=64): dbc = u @ xproj^T ; bf16 mirror fused
    k_gemm_wmma<<<dim3(64 / 16, Mn / 128), 256, 0, stream>>>(
        u_bf, w_x_bf, dbc, dbc_bf, En, En, En, 64, nullptr, 0);
    // GEMM3 (simple, K=32): dt = softplus(dbc[:, :32] @ dtw^T + dtb)
    k_gemm_wmma<<<dim3(En / 16, Mn / 128), 256, 0, stream>>>(
        dbc_bf, w_dt_bf, dtbuf, nullptr, Rn, 64, Rn, En, dtb, 1);
    // selective scan; writes y over u (f32) and into u_bf
    k_scan<<<Bn * 4, 256, 0, stream>>>(dtbuf, dbc, xz, u, u_bf, Alog, Dv);
    // GEMM4 (blocked): h[M,H] = y @ outw^T   (M=8192, N=512, K=1024); bf16 mirror fused
    k_gemm_wmma_blk<<<dim3(Hn / 64, Mn / 512), 256, 0, stream>>>(
        u_bf, w_ow_bf, h, h_bf, En, En, En, Hn);
  }

  k_head<<<Bn, 256, 0, stream>>>(h, lengths, out_w, out_b, out);
}
